// GAT_GCN_76304388980944
// MI455X (gfx1250) — matrix operand order repaired and checked by reading, not compiled, not run on hardware
//
#include <hip/hip_runtime.h>
#include <hip/hip_bf16.h>

#define N_NODES 8192
#define FEAT 256

typedef __attribute__((ext_vector_type(16))) __bf16 v16bf;
typedef __attribute__((ext_vector_type(8)))  float  v8f;

__device__ __forceinline__ __bf16 tobf(float f) { return (__bf16)f; }

// ---------------------------------------------------------------------------
// Kernel 1: H = X @ W   [8192,256] = [8192,256] x [256,256]  (bf16 WMMA)
// One wave per 16x16 output tile; K=256 in 8 steps of 32.
// ---------------------------------------------------------------------------
__global__ void __launch_bounds__(128)
k1_hprime(const float* __restrict__ X, const float* __restrict__ W,
          float* __restrict__ H) {
  const int lane = threadIdx.x & 31;
  const int wave = threadIdx.x >> 5;
  const int tile = blockIdx.x * 4 + wave;   // 8192 tiles = 512 x 16
  const int rb = (tile >> 4) * 16;
  const int nb = (tile & 15) * 16;
  const int m = lane & 15;
  const int h = lane >> 4;

  v8f acc = {};
  for (int kt = 0; kt < 8; ++kt) {
    const int kb = kt * 32;
    // A operand: lane holds row rb+m, K = kb + 8h + {0..7}  and  kb+16+8h + {0..7}
    const float* pa = X + (size_t)(rb + m) * FEAT + kb + h * 8;
    float4 a0 = *(const float4*)(pa);
    float4 a1 = *(const float4*)(pa + 4);
    float4 a2 = *(const float4*)(pa + 16);
    float4 a3 = *(const float4*)(pa + 20);
    v16bf A;
    A[0]=tobf(a0.x); A[1]=tobf(a0.y); A[2]=tobf(a0.z); A[3]=tobf(a0.w);
    A[4]=tobf(a1.x); A[5]=tobf(a1.y); A[6]=tobf(a1.z); A[7]=tobf(a1.w);
    A[8]=tobf(a2.x); A[9]=tobf(a2.y); A[10]=tobf(a2.z); A[11]=tobf(a2.w);
    A[12]=tobf(a3.x); A[13]=tobf(a3.y); A[14]=tobf(a3.z); A[15]=tobf(a3.w);
    // B operand: lane holds col nb+m, K = kb + 8h + {0..7}  and  kb+16+8h + {0..7}
    v16bf B;
    const float* pb = W + (size_t)(kb + h * 8) * FEAT + nb + m;
#pragma unroll
    for (int e = 0; e < 16; ++e) B[e] = tobf(pb[(size_t)((e & 7) + (e >> 3) * 16) * FEAT]);
    acc = __builtin_amdgcn_wmma_f32_16x16x32_bf16(false, A, false, B,
                                                  (short)0, acc, false, false);
  }
#pragma unroll
  for (int v = 0; v < 8; ++v)
    H[(size_t)(rb + v + 8 * h) * FEAT + nb + m] = acc[v];
}

// ---------------------------------------------------------------------------
// Kernel T: Vt[n][j] = bf16(H[j][n])   (LDS transpose, coalesced both ways)
// ---------------------------------------------------------------------------
__global__ void __launch_bounds__(256)
kT_transpose(const float* __restrict__ H, __bf16* __restrict__ Vt) {
  __shared__ float tile[32][33];
  const int tx = threadIdx.x & 31;
  const int ty = threadIdx.x >> 5;       // 0..7
  const int cb = blockIdx.x * 32;        // column of H (feature), 8 blocks
  const int rb = blockIdx.y * 32;        // row of H (node), 256 blocks
#pragma unroll
  for (int k = 0; k < 4; ++k)
    tile[ty + 8 * k][tx] = H[(size_t)(rb + ty + 8 * k) * FEAT + cb + tx];
  __syncthreads();
#pragma unroll
  for (int k = 0; k < 4; ++k)
    Vt[(size_t)(cb + ty + 8 * k) * N_NODES + rb + tx] = tobf(tile[tx][ty + 8 * k]);
}

// ---------------------------------------------------------------------------
// Kernel 2a: f_dst[i] = H[i,:] . q,  q = w_a @ a[2:4].  One wave per row.
// ---------------------------------------------------------------------------
__global__ void __launch_bounds__(256)
k2a_fdst(const float* __restrict__ H, const float* __restrict__ Wa,
         const float* __restrict__ Av, float* __restrict__ fdst) {
  __shared__ float q[FEAT];
  const int tid = threadIdx.x;
  q[tid] = Wa[2 * tid] * Av[2] + Wa[2 * tid + 1] * Av[3];
  __syncthreads();
  const int lane = tid & 31;
  const int wave = tid >> 5;
  const int row = blockIdx.x * 8 + wave;
  const float* p = H + (size_t)row * FEAT + lane * 8;
  float4 x0 = *(const float4*)(p);
  float4 x1 = *(const float4*)(p + 4);
  const float* ql = q + lane * 8;
  float s = x0.x * ql[0] + x0.y * ql[1] + x0.z * ql[2] + x0.w * ql[3]
          + x1.x * ql[4] + x1.y * ql[5] + x1.z * ql[6] + x1.w * ql[7];
#pragma unroll
  for (int off = 16; off; off >>= 1) s += __shfl_xor(s, off, 32);
  if (lane == 0) fdst[row] = s;
}

// ---------------------------------------------------------------------------
// Kernel 2b: M = max(f_dst); E[j] = exp(f_dst[j] - M)
// ---------------------------------------------------------------------------
__global__ void __launch_bounds__(256)
k2b_expw(const float* __restrict__ fdst, float* __restrict__ E) {
  __shared__ float red[256];
  const int tid = threadIdx.x;
  float mx = -1e30f;
  for (int i = tid; i < N_NODES; i += 256) mx = fmaxf(mx, fdst[i]);
  red[tid] = mx;
  __syncthreads();
  for (int s = 128; s; s >>= 1) {
    if (tid < s) red[tid] = fmaxf(red[tid], red[tid + s]);
    __syncthreads();
  }
  const float M = red[0];
  for (int i = tid; i < N_NODES; i += 256) E[i] = __expf(fdst[i] - M);
}

// ---------------------------------------------------------------------------
// Kernel 3: fused  mask*exp-weights -> S@V (WMMA) -> /Z -> ReLU
// Block: 16 rows x 256 cols of Out; 4 waves, wave w owns cols [64w, 64w+64).
// Ahat is binary, so weight = Ahat * E (no compare/select).
// Z is computed by a 5th WMMA vs an all-ones B matrix: accz element v holds
// Z_{row v+8h} in exactly the C/D layout -> no cross-lane shuffles needed,
// and normalization uses the same bf16-rounded weights as the matmul.
// ---------------------------------------------------------------------------
__global__ void __launch_bounds__(128)
k3_attn(const float* __restrict__ Ahat, const __bf16* __restrict__ Vt,
        const float* __restrict__ E, float* __restrict__ Out) {
  __shared__ float Es[N_NODES];          // 32 KB of the 320 KB WGP LDS
  const int tid = threadIdx.x;
  for (int i = tid; i < N_NODES / 4; i += 128)
    ((float4*)Es)[i] = ((const float4*)E)[i];
  __syncthreads();

  const int lane = tid & 31;
  const int wave = tid >> 5;
  const int rb = blockIdx.x * 16;
  const int m = lane & 15;
  const int h = lane >> 4;
  const int nbase = wave * 64;

  v8f acc[4];
#pragma unroll
  for (int t = 0; t < 4; ++t) acc[t] = (v8f){};
  v8f accz = {};

  v16bf ones;
#pragma unroll
  for (int e = 0; e < 16; ++e) ones[e] = tobf(1.0f);

  const float* arow = Ahat + (size_t)(rb + m) * N_NODES;

  for (int jb = 0; jb < N_NODES; jb += 32) {
    // prefetch the Ahat stream ~4KB ahead (wrapped in-row: always in-bounds)
    __builtin_prefetch(arow + ((jb + 1024) & (N_NODES - 1)), 0, 3);

    const int o1 = jb + h * 16;          // A-operand K layout: {16h..16h+7}
    const int o2 = o1 + 8;               //                 and {16h+8..16h+15}
    float4 a0 = *(const float4*)(arow + o1);
    float4 a1 = *(const float4*)(arow + o1 + 4);
    float4 a2 = *(const float4*)(arow + o2);
    float4 a3 = *(const float4*)(arow + o2 + 4);
    float4 e0 = *(const float4*)(Es + o1);
    float4 e1 = *(const float4*)(Es + o1 + 4);
    float4 e2 = *(const float4*)(Es + o2);
    float4 e3 = *(const float4*)(Es + o2 + 4);
    // Ahat is exactly 0.0 or 1.0: masked weight = A * E
    v16bf S;
    S[0]  = tobf(a0.x * e0.x);  S[1]  = tobf(a0.y * e0.y);
    S[2]  = tobf(a0.z * e0.z);  S[3]  = tobf(a0.w * e0.w);
    S[4]  = tobf(a1.x * e1.x);  S[5]  = tobf(a1.y * e1.y);
    S[6]  = tobf(a1.z * e1.z);  S[7]  = tobf(a1.w * e1.w);
    S[8]  = tobf(a2.x * e2.x);  S[9]  = tobf(a2.y * e2.y);
    S[10] = tobf(a2.z * e2.z);  S[11] = tobf(a2.w * e2.w);
    S[12] = tobf(a3.x * e3.x);  S[13] = tobf(a3.y * e3.y);
    S[14] = tobf(a3.z * e3.z);  S[15] = tobf(a3.w * e3.w);

    // Z row-sums ride the matrix pipe; lands in C/D layout directly.
    accz = __builtin_amdgcn_wmma_f32_16x16x32_bf16(false, S, false, ones,
                                                   (short)0, accz, false, false);
#pragma unroll
    for (int t = 0; t < 4; ++t) {
      // B operand: lane holds col nbase+16t+m, K = jb + 16h + {0..15} contiguous
      const __bf16* pv = Vt + (size_t)(nbase + 16 * t + m) * N_NODES + jb + 16 * h;
      v16bf B = *(const v16bf*)pv;       // 32B aligned
      acc[t] = __builtin_amdgcn_wmma_f32_16x16x32_bf16(false, S, false, B,
                                                       (short)0, acc[t], false, false);
    }
  }

#pragma unroll
  for (int v = 0; v < 8; ++v) {
    const float inv = 1.0f / accz[v];    // Z for row rb + v + 8h (any column)
    const int row = rb + v + 8 * h;
#pragma unroll
    for (int t = 0; t < 4; ++t) {
      float val = acc[t][v] * inv;
      Out[(size_t)row * FEAT + nbase + 16 * t + m] = fmaxf(val, 0.f);
    }
  }
}

// ---------------------------------------------------------------------------
extern "C" void kernel_launch(void* const* d_in, const int* in_sizes, int n_in,
                              void* d_out, int out_size, void* d_ws, size_t ws_size,
                              hipStream_t stream) {
  const float* X    = (const float*)d_in[0];   // node_feats [8192,256]
  const float* Ahat = (const float*)d_in[1];   // [8192,8192]
  const float* W    = (const float*)d_in[2];   // [256,256]
  const float* Wa   = (const float*)d_in[3];   // [256,2]
  const float* Av   = (const float*)d_in[4];   // [4,1]
  float* Out = (float*)d_out;                  // [8192,256]

  char* ws = (char*)d_ws;
  float*  H    = (float*)(ws);                                   // 8 MB
  __bf16* Vt   = (__bf16*)(ws + (8u << 20));                     // 4 MB
  float*  fdst = (float*)(ws + (8u << 20) + (4u << 20));         // 32 KB
  float*  E    = (float*)(ws + (8u << 20) + (4u << 20) + 32768); // 32 KB

  k1_hprime<<<2048, 128, 0, stream>>>(X, W, H);
  kT_transpose<<<dim3(8, 256), 256, 0, stream>>>(H, Vt);
  k2a_fdst<<<1024, 256, 0, stream>>>(H, Wa, Av, fdst);
  k2b_expw<<<1, 256, 0, stream>>>(fdst, E);
  k3_attn<<<512, 128, 0, stream>>>(Ahat, Vt, E, Out);
}
